// RelativeMultiHeadDotProductAttention_79010218377575
// MI455X (gfx1250) — compile-verified, matrix operand order checked by reading, not run on hardware
//
#include <hip/hip_runtime.h>
#include <hip/hip_bf16.h>

typedef __attribute__((ext_vector_type(16))) _Float16 v16h;
typedef __attribute__((ext_vector_type(8)))  _Float16 v8h;
typedef __attribute__((ext_vector_type(8)))  float    v8f;
typedef __attribute__((ext_vector_type(4)))  float    v4f;
typedef __attribute__((ext_vector_type(4)))  unsigned int uint4v;
typedef __attribute__((ext_vector_type(8)))  int      int8v;
typedef __attribute__((ext_vector_type(4)))  int      int4v;

#define S_LEN  1024
#define DMODEL 256
#define NHEAD  8
#define HDIM   32

static __device__ __forceinline__ v8f wmma16(v16h a, v16h b, v8f c) {
  // D = A(16x32 f16) x B(32x16 f16) + C(16x16 f32)
  return __builtin_amdgcn_wmma_f32_16x16x32_f16(false, a, false, b, (short)0, c,
                                                false, false);
}

// A/B fragment from f16 row-major [rows x ld]; lane holds row/col (lane&15),
// group g=lane>>4; element e -> K = g*8 + (e&7) + 16*(e>>3)
// => two contiguous 8-half (16B) runs per lane.
static __device__ __forceinline__ v16h load_frag_h(const _Float16* src, int ld,
                                                   int r0, int k0, int lane) {
  int r = r0 + (lane & 15);
  int g = lane >> 4;
  const _Float16* p = src + (long)r * ld + k0 + g * 8;
  v8h lo = *(const v8h*)(p);
  v8h hi = *(const v8h*)(p + 16);
  v16h f;
#pragma unroll
  for (int i = 0; i < 8; ++i) { f[i] = lo[i]; f[i + 8] = hi[i]; }
  return f;
}

// Same fragment from an f32 source with on-the-fly f16 conversion.
static __device__ __forceinline__ v16h load_frag_f32(const float* src, int ld,
                                                     int r0, int k0, int lane) {
  int r = r0 + (lane & 15);
  int g = lane >> 4;
  const float* p = src + (long)r * ld + k0 + g * 8;
  v4f a0 = *(const v4f*)(p);
  v4f a1 = *(const v4f*)(p + 4);
  v4f b0 = *(const v4f*)(p + 16);
  v4f b1 = *(const v4f*)(p + 20);
  v16h f;
#pragma unroll
  for (int i = 0; i < 4; ++i) {
    f[i]      = (_Float16)a0[i];
    f[i + 4]  = (_Float16)a1[i];
    f[i + 8]  = (_Float16)b0[i];
    f[i + 12] = (_Float16)b1[i];
  }
  return f;
}

// ---------------------------------------------------------------------------
// Stage `rows` x `rowlen` f16 (row-major, contiguous) global -> LDS.
// CDNA5: one TDM tensor_load_to_lds issued by wave 0, waited on TENSORcnt.
// D# per ISA 8.3/8.4: 2-D tensor, data_size=2B, type=2 ("image").
// ---------------------------------------------------------------------------
static __device__ __forceinline__ void stage_rows(_Float16* lds_dst,
                                                  const _Float16* gsrc,
                                                  int rows, int rowlen,
                                                  int tid, int nthreads) {
#if __has_builtin(__builtin_amdgcn_tensor_load_to_lds)
  if (tid < 32) {  // wave 0 issues the DMA (TDM ignores EXEC; branch is uniform)
    unsigned long long ga = (unsigned long long)(size_t)gsrc;
    unsigned lds_off = (unsigned)(size_t)lds_dst;  // low 32 bits = LDS byte addr
    uint4v g0;
    g0[0] = 1u;                                    // count=1, no gather/restore
    g0[1] = lds_off;                               // lds_addr
    g0[2] = (unsigned)ga;                          // global_addr[31:0]
    g0[3] = (unsigned)((ga >> 32) & 0x01FFFFFFu) | (2u << 30);  // [56:32]|type=2
    int8v g1;
    g1[0] = (int)(1u << 16);       // workgroup_mask=0, data_size=1 (2 bytes)
    g1[1] = rowlen << 16;          // tensor_dim0[15:0]  (bits 79:48 start @ b48)
    g1[2] = rows << 16;            // tensor_dim0[31:16]=0 | tensor_dim1[15:0]
    g1[3] = rowlen << 16;          // tensor_dim1[31:16]=0 | tile_dim0
    g1[4] = rows;                  // tile_dim1 | tile_dim2=0
    g1[5] = rowlen;                // tensor_dim0_stride[31:0]
    g1[6] = 0;                     // stride0[47:32] | tensor_dim1_stride[15:0]
    g1[7] = 0;
    int4v gz = {};
#if defined(__clang_major__) && __clang_major__ >= 23
    int8v gz8 = {};
    __builtin_amdgcn_tensor_load_to_lds(g0, g1, gz, gz, gz8, 0);
#else
    __builtin_amdgcn_tensor_load_to_lds(g0, g1, gz, gz, 0);
#endif
    __builtin_amdgcn_s_wait_tensorcnt(0);
  }
#else
  int total = rows * rowlen / 8;
  v8h* d = (v8h*)lds_dst;
  const v8h* s = (const v8h*)gsrc;
  for (int i = tid; i < total; i += nthreads) d[i] = s[i];
#endif
}

// ---------------------------------------------------------------------------
// Kernel 1: transpose weights to f16 [N,K] row-major (B-operand layout).
// Wq/Wk/Wv flat: [D=256][NH*HD=256]; Wo flat: [NH*HD=256][D=256].
// ---------------------------------------------------------------------------
__global__ void prep_weights_k(const float* Wq, const float* Wk, const float* Wv,
                               const float* Wo, _Float16* WqT, _Float16* WkT,
                               _Float16* WvT, _Float16* WoT) {
  int idx = blockIdx.x * blockDim.x + threadIdx.x;   // 0..65535
  int n = idx >> 8;
  int k = idx & 255;
  WqT[n * 256 + k] = (_Float16)Wq[k * 256 + n];
  WkT[n * 256 + k] = (_Float16)Wk[k * 256 + n];
  WvT[n * 256 + k] = (_Float16)Wv[k * 256 + n];
  WoT[n * 256 + k] = (_Float16)Wo[k * 256 + n];
}

// ---------------------------------------------------------------------------
// Kernel 2: fused QKV projection. Per block: one (which,b,n0) slice; weight
// rows n0..n0+31 staged to LDS via TDM and shared by 8 waves. Each wave: one
// 32x32 output tile = 2x2 accumulators, 4 WMMAs per K-step (fragment reuse).
// Q,K stored f16 [b][h][s][hd]; V stored transposed [b][h][hd][s].
// ---------------------------------------------------------------------------
__global__ void qkv_proj_k(const float* inq, const float* inkv,
                           const _Float16* WqT, const _Float16* WkT,
                           const _Float16* WvT, const float* bq, const float* bk,
                           const float* bv, _Float16* Q, _Float16* K,
                           _Float16* Vt) {
  __shared__ __align__(16) _Float16 w_s[32 * 256];   // 16 KB weight slice
  int lane = threadIdx.x & 31;
  int wave = threadIdx.x >> 5;
  int blk = blockIdx.x;
  int nt = blk & 7;                // n0 = nt*32
  int mg = (blk >> 3) & 3;
  int b  = (blk >> 5) & 7;
  int which = blk >> 8;            // 0=Q 1=K 2=V
  int n0 = nt * 32;
  int m0 = (mg * 8 + wave) * 32;

  const _Float16* WT = (which == 0) ? WqT : (which == 1 ? WkT : WvT);
  stage_rows(w_s, WT + (long)n0 * DMODEL, 32, DMODEL, threadIdx.x, 256);
  __syncthreads();

  const float* A = (which == 0 ? inq : inkv) + (long)b * S_LEN * DMODEL;
  const float* bias = (which == 0) ? bq : (which == 1 ? bk : bv);

  v8f a00 = {}, a01 = {}, a10 = {}, a11 = {};
#pragma unroll
  for (int k0 = 0; k0 < DMODEL; k0 += 32) {
    v16h x0 = load_frag_f32(A, DMODEL, m0,      k0, lane);
    v16h x1 = load_frag_f32(A, DMODEL, m0 + 16, k0, lane);
    v16h w0 = load_frag_h(w_s, DMODEL, 0,  k0, lane);
    v16h w1 = load_frag_h(w_s, DMODEL, 16, k0, lane);
    a00 = wmma16(x0, w0, a00);
    a01 = wmma16(x0, w1, a01);
    a10 = wmma16(x1, w0, a10);
    a11 = wmma16(x1, w1, a11);
  }

  int col = lane & 15;
  int rbase = (lane >> 4) * 8;
  float scale = (which == 0) ? 0.17677669529663687f : 1.0f;  // 1/sqrt(HD)
  float bn0 = bias[n0 + col];
  float bn1 = bias[n0 + 16 + col];
#pragma unroll
  for (int q = 0; q < 4; ++q) {
    v8f acc = (q == 0) ? a00 : (q == 1) ? a01 : (q == 2) ? a10 : a11;
    int n = n0 + ((q & 1) ? 16 : 0) + col;
    float bn = (q & 1) ? bn1 : bn0;
    int h = n >> 5, hd = n & 31;
#pragma unroll
    for (int r = 0; r < 8; ++r) {
      int m = m0 + ((q >> 1) ? 16 : 0) + rbase + r;
      float val = (acc[r] + bn) * scale;
      if (which == 0)
        Q[((long)(b * NHEAD + h) * S_LEN + m) * HDIM + hd] = (_Float16)val;
      else if (which == 1)
        K[((long)(b * NHEAD + h) * S_LEN + m) * HDIM + hd] = (_Float16)val;
      else
        Vt[((long)(b * NHEAD + h) * HDIM + hd) * S_LEN + m] = (_Float16)val;
    }
  }
}

// ---------------------------------------------------------------------------
// Online-softmax update for one 16-row tile against 32 new keys (s0|s1),
// rescale running output (o0|o1), write P (f16) into the wave's LDS tile.
// ---------------------------------------------------------------------------
static __device__ __forceinline__ void softmax_step(v8f& s0, v8f& s1, v8f& o0,
                                                    v8f& o1, float* mrow,
                                                    float* lrow, _Float16* prow,
                                                    int col, int rbase) {
#pragma unroll
  for (int r = 0; r < 8; ++r) {
    float mx = fmaxf(s0[r], s1[r]);
#pragma unroll
    for (int d = 1; d < 16; d <<= 1) mx = fmaxf(mx, __shfl_xor(mx, d, 16));
    float mnew = fmaxf(mrow[r], mx);
    float corr = __expf(mrow[r] - mnew);
    float p0 = __expf(s0[r] - mnew);
    float p1 = __expf(s1[r] - mnew);
    float rs = p0 + p1;
#pragma unroll
    for (int d = 1; d < 16; d <<= 1) rs += __shfl_xor(rs, d, 16);
    lrow[r] = lrow[r] * corr + rs;
    mrow[r] = mnew;
    o0[r] *= corr;
    o1[r] *= corr;
    prow[(rbase + r) * 32 + col]      = (_Float16)p0;
    prow[(rbase + r) * 32 + 16 + col] = (_Float16)p1;
  }
}

// ---------------------------------------------------------------------------
// Kernel 3: flash attention per (b,h). One wave owns 32 query rows; streams
// keys 32 at a time: 4 WMMAs (QK^T, K/V fragments reused across both query
// tiles) + analytic Swin rel-pos bias + online softmax + LDS repack of P +
// 4 WMMAs (P.V^T). Output X f16 [b][s][h*HD+hd].
// ---------------------------------------------------------------------------
__global__ void attn_k(const _Float16* Q, const _Float16* K, const _Float16* Vt,
                       const float* rel, _Float16* X) {
  __shared__ __align__(16) _Float16 pbuf[8][32 * 32];  // per-wave P tile (2KB)
  int lane = threadIdx.x & 31;
  int wave = threadIdx.x >> 5;
  int blk = blockIdx.x;
  int qt = (blk & 3) * 8 + wave;     // 32-row query tile, 0..31
  int bh = blk >> 2;                 // (b*NHEAD + h)
  int h = bh & 7, b = bh >> 3;

  const _Float16* Qp  = Q  + (long)bh * S_LEN * HDIM;
  const _Float16* Kp  = K  + (long)bh * S_LEN * HDIM;
  const _Float16* Vtp = Vt + (long)bh * HDIM * S_LEN;
  _Float16* pw = &pbuf[wave][0];

  int m0 = qt * 32;
  v16h qa0 = load_frag_h(Qp, HDIM, m0,      0, lane);  // full HD in one frag
  v16h qa1 = load_frag_h(Qp, HDIM, m0 + 16, 0, lane);

  v8f o00 = {}, o01 = {}, o10 = {}, o11 = {};
  float m0r[8], l0r[8], m1r[8], l1r[8];
#pragma unroll
  for (int r = 0; r < 8; ++r) {
    m0r[r] = -3.0e38f; l0r[r] = 0.0f;
    m1r[r] = -3.0e38f; l1r[r] = 0.0f;
  }

  int col = lane & 15;
  int rbase = (lane >> 4) * 8;

  for (int j0 = 0; j0 < S_LEN; j0 += 32) {
    if (j0 + 32 < S_LEN) {
      __builtin_prefetch(Kp  + (long)(j0 + 32) * HDIM, 0, 1);
      __builtin_prefetch(Vtp + (long)(j0 + 32), 0, 1);
    }
    v16h kb0 = load_frag_h(Kp, HDIM, j0,      0, lane);
    v16h kb1 = load_frag_h(Kp, HDIM, j0 + 16, 0, lane);
    v8f z = {};
    v8f s00 = wmma16(qa0, kb0, z);
    v8f s01 = wmma16(qa0, kb1, z);
    v8f s10 = wmma16(qa1, kb0, z);
    v8f s11 = wmma16(qa1, kb1, z);

    // Swin relative-position bias (POS table is an analytic constant).
    int k0i = j0 + col,      k1i = j0 + 16 + col;
    int yk0 = k0i >> 5, xk0 = k0i & 31;
    int yk1 = k1i >> 5, xk1 = k1i & 31;
#pragma unroll
    for (int r = 0; r < 8; ++r) {
      int qi0 = m0 + rbase + r;
      int qi1 = qi0 + 16;
      int yq0 = qi0 >> 5, xq0 = qi0 & 31;
      int yq1 = qi1 >> 5, xq1 = qi1 & 31;
      s00[r] += rel[((yq0 - yk0 + 31) * 63 + (xq0 - xk0 + 31)) * NHEAD + h];
      s01[r] += rel[((yq0 - yk1 + 31) * 63 + (xq0 - xk1 + 31)) * NHEAD + h];
      s10[r] += rel[((yq1 - yk0 + 31) * 63 + (xq1 - xk0 + 31)) * NHEAD + h];
      s11[r] += rel[((yq1 - yk1 + 31) * 63 + (xq1 - xk1 + 31)) * NHEAD + h];
    }

    softmax_step(s00, s01, o00, o01, m0r, l0r, pw,       col, rbase);
    softmax_step(s10, s11, o10, o11, m1r, l1r, pw + 512, col, rbase);

    // Repack P (C layout -> A layout) through LDS, then P x V^T.
    v16h pa0 = load_frag_h(pw, 32, 0,  0, lane);
    v16h pa1 = load_frag_h(pw, 32, 16, 0, lane);
    v16h vb0 = load_frag_h(Vtp, S_LEN, 0,  j0, lane);
    v16h vb1 = load_frag_h(Vtp, S_LEN, 16, j0, lane);
    o00 = wmma16(pa0, vb0, o00);
    o01 = wmma16(pa0, vb1, o01);
    o10 = wmma16(pa1, vb0, o10);
    o11 = wmma16(pa1, vb1, o11);
  }

#pragma unroll
  for (int r = 0; r < 8; ++r) {
    float inv0 = 1.0f / l0r[r];
    float inv1 = 1.0f / l1r[r];
    int s0i = m0 + rbase + r;
    int s1i = s0i + 16;
    long base0 = ((long)b * S_LEN + s0i) * DMODEL + h * HDIM;
    long base1 = ((long)b * S_LEN + s1i) * DMODEL + h * HDIM;
    X[base0 + col]      = (_Float16)(o00[r] * inv0);
    X[base0 + 16 + col] = (_Float16)(o01[r] * inv0);
    X[base1 + col]      = (_Float16)(o10[r] * inv1);
    X[base1 + 16 + col] = (_Float16)(o11[r] * inv1);
  }
}

// ---------------------------------------------------------------------------
// Kernel 4: output projection [8192 x 256] x [256 x 256] + bo -> f32 out.
// Per block: 32 output columns; WoT rows n0..n0+31 staged to LDS via TDM and
// shared by 8 waves; each wave one 32x32 tile (2x2 accumulators).
// ---------------------------------------------------------------------------
__global__ void out_proj_k(const _Float16* X, const _Float16* WoT,
                           const float* bo, float* out) {
  __shared__ __align__(16) _Float16 wo_s[32 * 256];  // 16 KB
  int lane = threadIdx.x & 31;
  int wave = threadIdx.x >> 5;
  int nb = blockIdx.x & 7;           // n0 = nb*32
  int mg = blockIdx.x >> 3;          // 0..31
  int n0 = nb * 32;
  int m0 = (mg * 8 + wave) * 32;

  stage_rows(wo_s, WoT + (long)n0 * DMODEL, 32, DMODEL, threadIdx.x, 256);
  __syncthreads();

  v8f a00 = {}, a01 = {}, a10 = {}, a11 = {};
#pragma unroll
  for (int k0 = 0; k0 < DMODEL; k0 += 32) {
    v16h x0 = load_frag_h(X, DMODEL, m0,      k0, lane);
    v16h x1 = load_frag_h(X, DMODEL, m0 + 16, k0, lane);
    v16h w0 = load_frag_h(wo_s, DMODEL, 0,  k0, lane);
    v16h w1 = load_frag_h(wo_s, DMODEL, 16, k0, lane);
    a00 = wmma16(x0, w0, a00);
    a01 = wmma16(x0, w1, a01);
    a10 = wmma16(x1, w0, a10);
    a11 = wmma16(x1, w1, a11);
  }

  int col = lane & 15;
  int rbase = (lane >> 4) * 8;
  float bn0 = bo[n0 + col];
  float bn1 = bo[n0 + 16 + col];
#pragma unroll
  for (int r = 0; r < 8; ++r) {
    int ma = m0 + rbase + r;
    int mb2 = ma + 16;
    out[(long)ma  * DMODEL + n0 + col]      = a00[r] + bn0;
    out[(long)ma  * DMODEL + n0 + 16 + col] = a01[r] + bn1;
    out[(long)mb2 * DMODEL + n0 + col]      = a10[r] + bn0;
    out[(long)mb2 * DMODEL + n0 + 16 + col] = a11[r] + bn1;
  }
}

extern "C" void kernel_launch(void* const* d_in, const int* in_sizes, int n_in,
                              void* d_out, int out_size, void* d_ws, size_t ws_size,
                              hipStream_t stream) {
  (void)in_sizes; (void)n_in; (void)out_size; (void)ws_size;
  const float* inq  = (const float*)d_in[0];
  const float* inkv = (const float*)d_in[1];
  const float* Wq   = (const float*)d_in[2];
  const float* bq   = (const float*)d_in[3];
  const float* Wk   = (const float*)d_in[4];
  const float* bk   = (const float*)d_in[5];
  const float* Wv   = (const float*)d_in[6];
  const float* bv   = (const float*)d_in[7];
  const float* rel  = (const float*)d_in[8];
  const float* Wo   = (const float*)d_in[9];
  const float* bo   = (const float*)d_in[10];
  float* out = (float*)d_out;

  char* ws = (char*)d_ws;
  const size_t WSZ = 256 * 256 * sizeof(_Float16);                 // 128 KB
  const size_t QSZ = (size_t)8 * 8 * 1024 * 32 * sizeof(_Float16); // 4 MB
  _Float16* WqT = (_Float16*)(ws + 0 * WSZ);
  _Float16* WkT = (_Float16*)(ws + 1 * WSZ);
  _Float16* WvT = (_Float16*)(ws + 2 * WSZ);
  _Float16* WoT = (_Float16*)(ws + 3 * WSZ);
  size_t off = 4 * WSZ;
  _Float16* Qb = (_Float16*)(ws + off); off += QSZ;
  _Float16* Kb = (_Float16*)(ws + off); off += QSZ;
  _Float16* Vt = (_Float16*)(ws + off); off += QSZ;
  _Float16* Xb = (_Float16*)(ws + off); off += QSZ;

  prep_weights_k<<<256, 256, 0, stream>>>(Wq, Wk, Wv, Wo, WqT, WkT, WvT, WoT);
  qkv_proj_k<<<768, 256, 0, stream>>>(inq, inkv, WqT, WkT, WvT, bq, bk, bv,
                                      Qb, Kb, Vt);
  attn_k<<<256, 256, 0, stream>>>(Qb, Kb, Vt, rel, Xb);
  out_proj_k<<<256, 256, 0, stream>>>(Xb, WoT, bo, out);
}
